// MahalanobisBlock_43404939493451
// MI455X (gfx1250) — matrix operand order repaired and checked by reading, not compiled, not run on hardware
//
#include <hip/hip_runtime.h>

// ---------------------------------------------------------------------------
// Mahalanobis similarity block for MI455X (gfx1250, wave32, WMMA, TDM)
//   x1: [64,256,21,21] f32 queries      x2: [16,8,256,21,21] f32 supports
//   out: [64, 16*441] f32
// Pipeline: colsum -> qprep(diff bf16) -> Gram+finalize(cov bf16, WMMA)
//           -> sim = diag(diffT cov diff) via WMMA, cov tiles streamed by
//              the Tensor Data Mover (double-buffered, TENSORcnt-synced)
// ---------------------------------------------------------------------------

#define B_   64
#define N_   16
#define BS_  8
#define C_   256
#define HW_  441
#define HWP_ 448            // HW padded to multiple of 16
#define R_   3528           // BS_*HW_
#define CHW_ 112896         // C_*HW_ (stride between support samples)

typedef __attribute__((ext_vector_type(16))) __bf16          v16bf;
typedef __attribute__((ext_vector_type(16))) unsigned short  v16u;
typedef __attribute__((ext_vector_type(8)))  float           v8f;
typedef __attribute__((ext_vector_type(4)))  unsigned int    u32x4;
typedef __attribute__((ext_vector_type(8)))  int             i32x8;
typedef __attribute__((ext_vector_type(4)))  int             i32x4;

#if __has_builtin(__builtin_amdgcn_tensor_load_to_lds)
#define HAVE_TDM 1
#else
#define HAVE_TDM 0
#endif

__device__ __forceinline__ unsigned short f2bf(float f) {
  unsigned int u = __builtin_bit_cast(unsigned int, f);
  u += 0x7FFFu + ((u >> 16) & 1u);               // round-to-nearest-even
  return (unsigned short)(u >> 16);
}
__device__ __forceinline__ float bf2f(unsigned short h) {
  unsigned int u = ((unsigned int)h) << 16;
  return __builtin_bit_cast(float, u);
}
__device__ __forceinline__ v8f zero8() {
  v8f z = {0.f, 0.f, 0.f, 0.f, 0.f, 0.f, 0.f, 0.f};
  return z;
}

// LDS tile layout: [idx16][36] bf16, K contiguous (pad 36 => 72B row stride:
// 32b-aligned pair loads, 18-bank stride => conflict-light gathers).
// A fragment (16x32 MxK): lane l -> m=l&15; K set {koff..koff+15}
//   with koff = (l>=16)?16:0
__device__ __forceinline__ v16bf frag_A(const unsigned short* lds, int row16, int lane) {
  const int m    = lane & 15;
  const int koff = (lane >> 4) << 4;
  const unsigned short* rp = lds + (row16 + m) * 36;
  v16u u;
#pragma unroll
  for (int v = 0; v < 4; ++v) {
    unsigned int w0 = *(const unsigned int*)(rp + koff + 2 * v);
    u[2 * v]     = (unsigned short)(w0 & 0xFFFFu);
    u[2 * v + 1] = (unsigned short)(w0 >> 16);
    unsigned int w1 = *(const unsigned int*)(rp + 8 + koff + 2 * v);
    u[8 + 2 * v]     = (unsigned short)(w1 & 0xFFFFu);
    u[8 + 2 * v + 1] = (unsigned short)(w1 >> 16);
  }
  return __builtin_bit_cast(v16bf, u);
}
// B fragment (32x16 KxN): lane l -> n=l&15; K half {0..15} or {16..31}
__device__ __forceinline__ v16bf frag_B(const unsigned short* lds, int col16, int lane) {
  const int nn   = lane & 15;
  const int koff = (lane >> 4) << 4;
  const unsigned short* cp = lds + (col16 + nn) * 36 + koff;
  v16u u;
#pragma unroll
  for (int v = 0; v < 8; ++v) {
    unsigned int w0 = *(const unsigned int*)(cp + 2 * v);
    u[2 * v]     = (unsigned short)(w0 & 0xFFFFu);
    u[2 * v + 1] = (unsigned short)(w0 >> 16);
  }
  return __builtin_bit_cast(v16bf, u);
}

#if HAVE_TDM
// TDM: DMA a [256 rows x 32 bf16] tile (row stride 256 elems) into LDS with
// hardware padding 64B-row -> 72B-row (pad_interval=16 DWORDs, pad_amount=2
// DWORDs), reproducing the [row][36] gather layout. D# per ISA 8.3/8.4.
// This toolchain's builtin is the 6-arg form:
//   (uint32x4 g0, int32x8 g1, int32x4 g2, int32x4 g3, int32x8 g4, i32 cpol)
__device__ __forceinline__ void tdm_load_cov(const unsigned short* gsrc,
                                             unsigned int lds_off) {
  unsigned long long ga = (unsigned long long)(uintptr_t)gsrc;
  u32x4 g0;
  g0[0] = 1u;                                            // count=1, user mode
  g0[1] = lds_off;                                       // D#.lds_addr
  g0[2] = (unsigned int)ga;                              // global_addr[31:0]
  g0[3] = (unsigned int)((ga >> 32) & 0x01FFFFFFull)     // global_addr[56:32]
          | 0x80000000u;                                 // type=2 ("image")
  i32x8 g1;
  g1[0] = (1 << 16)       // data_size = 2 bytes
        | (1 << 20)       // pad_enable
        | (3 << 22)       // pad_interval: 16 DWORDs (= 64B row)
        | (1 << 25);      // pad_amount: 2 DWORDs (= 4 bf16 -> 36-elem rows)
  g1[1] = (int)(32u << 16);   // tensor_dim0 = 32   (bits 48..79, low half)
  g1[2] = (int)(256u << 16);  // tensor_dim1 = 256  (bits 80..111, low half)
  g1[3] = (int)(32u << 16);   // tile_dim0 = 32     (bits 112..127)
  g1[4] = 256;                // tile_dim1 = 256    (bits 128..143)
  g1[5] = 256;                // tensor_dim0_stride = 256 (bits 160..191)
  g1[6] = 0;
  g1[7] = 0;
  i32x4 z4 = {0, 0, 0, 0};
  i32x8 z8 = {0, 0, 0, 0, 0, 0, 0, 0};
  __builtin_amdgcn_tensor_load_to_lds(g0, g1, z4, z4, z8, 0);
}
#endif

// ---------------------------------------------------------------------------
// Kernel 1: per-class per-channel column sums over the R=3528 support rows
// ---------------------------------------------------------------------------
__global__ void __launch_bounds__(256) colsum_kernel(const float* __restrict__ x2,
                                                     float* __restrict__ colsum) {
  const int nc = blockIdx.x;               // N_*C_ = 4096
  const int n = nc >> 8, c = nc & 255;
  const float* base = x2 + (size_t)n * BS_ * CHW_ + (size_t)c * HW_;
  float s = 0.f;
  for (int i = threadIdx.x; i < R_; i += 256) {
    int bs = i / HW_, p = i - bs * HW_;
    s += base[(size_t)bs * CHW_ + p];
  }
  __shared__ float red[256];
  red[threadIdx.x] = s;
  __syncthreads();
  for (int off = 128; off >= 1; off >>= 1) {
    if (threadIdx.x < (unsigned)off) red[threadIdx.x] += red[threadIdx.x + off];
    __syncthreads();
  }
  if (threadIdx.x == 0) colsum[nc] = red[0];
}

// ---------------------------------------------------------------------------
// Kernel 2: query prep: diff = (x - mean(x)) / ||x||, bf16, HW padded to 448
// ---------------------------------------------------------------------------
__global__ void __launch_bounds__(256) qprep_kernel(const float* __restrict__ x1,
                                                    unsigned short* __restrict__ diff) {
  const int bc = blockIdx.x;               // B_*C_ = 16384
  const float* xp = x1 + (size_t)bc * HW_;
  unsigned short* dp = diff + (size_t)bc * HWP_;
  float s = 0.f, sq = 0.f;
  for (int i = threadIdx.x; i < HW_; i += 256) {
    float v = xp[i];
    s += v;
    sq += v * v;
  }
  __shared__ float rs[256], rq[256];
  rs[threadIdx.x] = s;
  rq[threadIdx.x] = sq;
  __syncthreads();
  for (int off = 128; off >= 1; off >>= 1) {
    if (threadIdx.x < (unsigned)off) {
      rs[threadIdx.x] += rs[threadIdx.x + off];
      rq[threadIdx.x] += rq[threadIdx.x + off];
    }
    __syncthreads();
  }
  const float inv  = rsqrtf(rq[0]);
  const float mean = rs[0] * (1.0f / (float)HW_);
  for (int i = threadIdx.x; i < HWP_; i += 256) {
    float v = (i < HW_) ? (xp[i] - mean) * inv : 0.f;
    dp[i] = f2bf(v);
  }
}

// ---------------------------------------------------------------------------
// Kernel 3: Gram GEMM per class + covariance finalize -> bf16 cov
//   grid: 16 classes x 16 (4x4) 64x64 output tiles; 128 thr = 4 waves
//   wave w: output rows [16w,16w+16), 4 col-tiles -> 4 accumulators
// ---------------------------------------------------------------------------
__global__ void __launch_bounds__(128) cov_kernel(const float* __restrict__ x2,
                                                  const float* __restrict__ colsum,
                                                  unsigned short* __restrict__ covb) {
  const int wg = blockIdx.x;               // 256
  const int n = wg >> 4;
  const int ti = (wg >> 2) & 3, tj = wg & 3;
  const int ci0 = ti * 64, cj0 = tj * 64;
  __shared__ unsigned short XA[64 * 36];
  __shared__ unsigned short XB[64 * 36];
  const int t = threadIdx.x, lane = t & 31, w = t >> 5;
  const float* xn = x2 + (size_t)n * BS_ * CHW_;
  const int cl = t >> 1, rh = t & 1;       // 64 cols x 2 row-halves

  v8f acc[4];
#pragma unroll
  for (int j = 0; j < 4; ++j) acc[j] = zero8();

  for (int r0 = 0; r0 < 3552; r0 += 32) {  // 111 K-steps of 32 (zero-padded)
#pragma unroll 4
    for (int i = 0; i < 16; ++i) {
      int r = r0 + rh * 16 + i;
      float va = 0.f, vb = 0.f;
      if (r < R_) {
        int bs = r / HW_, p = r - bs * HW_;
        size_t o = (size_t)bs * CHW_ + p;
        va = xn[o + (size_t)(ci0 + cl) * HW_];
        vb = xn[o + (size_t)(cj0 + cl) * HW_];
      }
      XA[cl * 36 + rh * 16 + i] = f2bf(va);
      XB[cl * 36 + rh * 16 + i] = f2bf(vb);
    }
    __syncthreads();
    v16bf a = frag_A(XA, w * 16, lane);
#pragma unroll
    for (int j = 0; j < 4; ++j) {
      v16bf b = frag_B(XB, j * 16, lane);
      acc[j] = __builtin_amdgcn_wmma_f32_16x16x32_bf16(
          false, a, false, b, (short)0, acc[j], false, false);
    }
    __syncthreads();
  }

  // finalize: cov = (S - R*mu_r*mu_c) / (R-1), cast bf16
  const float invRm1 = 1.0f / (float)(R_ - 1);
  const float invR   = 1.0f / (float)R_;
  const int mlo = (lane >> 4) * 8;
  const int col_l = lane & 15;
#pragma unroll
  for (int j = 0; j < 4; ++j) {
    int col = cj0 + j * 16 + col_l;
    float mc = colsum[n * C_ + col] * invR;
#pragma unroll
    for (int v = 0; v < 8; ++v) {
      int row = ci0 + w * 16 + mlo + v;
      float mr = colsum[n * C_ + row] * invR;
      float cv = (acc[j][v] - (float)R_ * mr * mc) * invRm1;
      covb[(size_t)n * C_ * C_ + (size_t)row * C_ + col] = f2bf(cv);
    }
  }
}

// ---------------------------------------------------------------------------
// Kernel 4: sim[b,n,k] = sum_c diff[c,k] * (cov[n] @ diff)[c,k]
//   grid: B*N*7 workgroups; WG = 128 thr = 4 waves, each wave owns 16 cols
//   16 f32 accumulators/wave; cov K-slices (16 KB) are streamed into LDS by
//   the Tensor Data Mover, double-buffered, with s_wait_tensorcnt overlap.
// ---------------------------------------------------------------------------
__global__ void __launch_bounds__(128) sim_kernel(const unsigned short* __restrict__ covb,
                                                  const unsigned short* __restrict__ diffb,
                                                  float* __restrict__ out) {
  const int wg = blockIdx.x;               // 64*16*7 = 7168
  const int cb = wg % 7;
  const int bn = wg / 7;
  const int n = bn & 15, b = bn >> 4;
  const int col0 = cb * 64;
  __shared__ unsigned short CV[2][C_ * 36]; // 2 x 18432 B (ping-pong TDM dest)
  __shared__ unsigned short DT[64 * 36];    //     4608 B
  const int t = threadIdx.x, lane = t & 31, w = t >> 5;
  const unsigned short* covn  = covb + (size_t)n * C_ * C_;
  const unsigned short* diffB = diffb + (size_t)b * C_ * HWP_;

  v8f acc[16];
#pragma unroll
  for (int i = 0; i < 16; ++i) acc[i] = zero8();

#if HAVE_TDM
  if (t < 32)                              // wave 0 drives the TDM
    tdm_load_cov(covn, (unsigned int)(uintptr_t)(&CV[0][0]));
#endif

  for (int kk = 0; kk < 8; ++kk) {         // K over channels d, 32 each
    const int cur = kk & 1;
#if !HAVE_TDM
    // fallback staging of cov rows 0..255, k-slice [kk*32, kk*32+32)
    for (int rr = t; rr < C_; rr += 128) {
      const ushort4* src = (const ushort4*)(covn + (size_t)rr * C_ + kk * 32);
      ushort4* dst = (ushort4*)(&CV[cur][rr * 36]);
#pragma unroll
      for (int j2 = 0; j2 < 8; ++j2) dst[j2] = src[j2];
    }
#endif
    // stage diff tile: 32 k-rows x 64 cols, transposed into [col][k]
    for (int q = t; q < 512; q += 128) {
      int k = q >> 4, c4 = (q & 15) << 2;
      ushort4 vsrc = *(const ushort4*)(diffB + (size_t)(kk * 32 + k) * HWP_ + col0 + c4);
      DT[(c4 + 0) * 36 + k] = vsrc.x;
      DT[(c4 + 1) * 36 + k] = vsrc.y;
      DT[(c4 + 2) * 36 + k] = vsrc.z;
      DT[(c4 + 3) * 36 + k] = vsrc.w;
    }
#if HAVE_TDM
    if (t < 32) {
      if (kk < 7) {                        // prefetch next slice, then wait
        tdm_load_cov(covn + (size_t)(kk + 1) * 32,
                     (unsigned int)(uintptr_t)(&CV[cur ^ 1][0]));
        __builtin_amdgcn_s_wait_tensorcnt(1);   // current slice complete
      } else {
        __builtin_amdgcn_s_wait_tensorcnt(0);
      }
    }
#endif
    __syncthreads();                       // CV[cur] + DT visible to all waves
    v16bf bfr = frag_B(DT, w * 16, lane);
#pragma unroll
    for (int ci = 0; ci < 16; ++ci) {
      v16bf afr = frag_A(&CV[cur][0], ci * 16, lane);
      acc[ci] = __builtin_amdgcn_wmma_f32_16x16x32_bf16(
          false, afr, false, bfr, (short)0, acc[ci], false, false);
    }
    __syncthreads();                       // all waves done with CV[cur], DT
  }

  // epilogue: s[col] = sum_rows diff[row,col] * Y[row,col]; halves via shfl_xor
  const int col = col0 + w * 16 + (lane & 15);
  const int mlo = (lane >> 4) * 8;
  float s = 0.f;
#pragma unroll
  for (int ci = 0; ci < 16; ++ci) {
#pragma unroll
    for (int v = 0; v < 8; ++v) {
      float dv = bf2f(diffB[(size_t)(ci * 16 + mlo + v) * HWP_ + col]);
      s += dv * acc[ci][v];
    }
  }
  s += __shfl_xor(s, 16, 32);              // combine M halves (wave32)
  if (lane < 16 && col < HW_)
    out[(size_t)b * (N_ * HW_) + n * HW_ + col] = s;
}

// ---------------------------------------------------------------------------
extern "C" void kernel_launch(void* const* d_in, const int* in_sizes, int n_in,
                              void* d_out, int out_size, void* d_ws, size_t ws_size,
                              hipStream_t stream) {
  (void)in_sizes; (void)n_in; (void)out_size; (void)ws_size;
  const float* x1 = (const float*)d_in[0];
  const float* x2 = (const float*)d_in[1];
  float* out = (float*)d_out;

  char* ws = (char*)d_ws;
  float*          colsum = (float*)ws;                                   // 16 KB
  unsigned short* covb   = (unsigned short*)(ws + 16384);                // 2 MB
  unsigned short* diffb  = (unsigned short*)(ws + 16384 + 2097152);      // 14.7 MB

  colsum_kernel<<<N_ * C_, 256, 0, stream>>>(x2, colsum);
  qprep_kernel<<<B_ * C_, 256, 0, stream>>>(x1, diffb);
  cov_kernel<<<N_ * 16, 128, 0, stream>>>(x2, colsum, covb);
  sim_kernel<<<B_ * N_ * 7, 128, 0, stream>>>(covb, diffb, out);
}
